// LowRankLinearAttention_3650722201705
// MI455X (gfx1250) — matrix operand order repaired and checked by reading, not compiled, not run on hardware
//
#include <hip/hip_runtime.h>

#define BB 4
#define SS 4096
#define DD 1024
#define KK 64
#define CC 128
#define NC (SS / CC)   // 32 chunks per batch
#define BS (BB * SS)   // 16384 rows
#define NDIT (DD / 32) // 32 d-slices in projection GEMM

typedef __bf16 bf16_t;
typedef __attribute__((ext_vector_type(8)))  bf16_t v8bf;
typedef __attribute__((ext_vector_type(16))) bf16_t v16bf;
typedef __attribute__((ext_vector_type(8)))  float  v8f;
typedef __attribute__((ext_vector_type(4)))  unsigned int u32x4;
typedef __attribute__((ext_vector_type(8)))  int i32x8;
typedef __attribute__((ext_vector_type(4)))  int i32x4;

__device__ __forceinline__ unsigned short f2bf(float f) {
  union { float f; unsigned u; } v; v.f = f;
  unsigned r = v.u + 0x7FFFu + ((v.u >> 16) & 1u);   // round-to-nearest-even
  return (unsigned short)(r >> 16);
}
__device__ __forceinline__ float bf2f(unsigned short u) {
  union { unsigned u; float f; } v; v.u = ((unsigned)u) << 16;
  return v.f;
}

// A-fragment (16x32 bf16, M x K): lane L holds row M=L&15.
// ISA table: lanes 0-15: K 0..7 (v0..3), K 16..23 (v4..7); lanes 16-31: K 8..15, K 24..31.
__device__ __forceinline__ v16bf load_fragA(const unsigned short* base, int stride, int lane) {
  const int idx = lane & 15, half = lane >> 4;
  const unsigned short* p = base + idx * stride + 8 * half;
  v8bf lo = *reinterpret_cast<const v8bf*>(p);
  v8bf hi = *reinterpret_cast<const v8bf*>(p + 16);
  return __builtin_shufflevector(lo, hi, 0,1,2,3,4,5,6,7,8,9,10,11,12,13,14,15);
}

// B-fragment (32x16 bf16, K x N): lane L holds col N=L&15; lanes 0-15: K 0..7, K 16..23; lanes 16-31: K 8..15, K 24..31.
// Storage must be [N][K] (K contiguous).
__device__ __forceinline__ v16bf load_fragB(const unsigned short* base, int stride, int lane) {
  const int idx = lane & 15, half = lane >> 4;
  const unsigned short* p = base + idx * stride + 8 * half;
  v8bf lo = *reinterpret_cast<const v8bf*>(p);
  v8bf hi = *reinterpret_cast<const v8bf*>(p + 16);
  return __builtin_shufflevector(lo, hi, 0,1,2,3,4,5,6,7,8,9,10,11,12,13,14,15);
}

__device__ __forceinline__ v8f wmma_bf16(v16bf a, v16bf b, v8f c) {
  return __builtin_amdgcn_wmma_f32_16x16x32_bf16(false, a, false, b, (short)0, c, false, false);
}

// TDM: async-copy a [192 rows x 32 halfs] tile (cols d0..d0+31) of the contiguous
// [192][1024] bf16 weight tensor into LDS. D# built per CDNA5 ISA.
__device__ __forceinline__ void tdm_load_w(const unsigned short* gtile, unsigned lds_bytes) {
  unsigned long long ga = (unsigned long long)gtile;
  u32x4 g0;
  g0[0] = 1u;                                              // count=1, user mode
  g0[1] = lds_bytes;                                       // lds_addr
  g0[2] = (unsigned)(ga & 0xffffffffu);                    // global_addr[31:0]
  g0[3] = (unsigned)((ga >> 32) & 0x1ffffffu) | (2u << 30); // global_addr[56:32], type=2
  i32x8 g1;
  g1[0] = 1 << 16;                                         // data_size=1 (2 bytes)
  g1[1] = (int)((1024u & 0xffffu) << 16);                  // tensor_dim0 lo16 (=1024)
  g1[2] = (int)((1024u >> 16) | ((192u & 0xffffu) << 16)); // dim0 hi16 | tensor_dim1 lo16 (=192)
  g1[3] = (int)((192u >> 16) | (32u << 16));               // dim1 hi16 | tile_dim0=32
  g1[4] = 192;                                             // tile_dim1=192, tile_dim2=0
  g1[5] = 1024;                                            // tensor_dim0_stride lo32
  g1[6] = 0;                                               // stride hi16 | dim1_stride lo16
  g1[7] = 0;
  i32x4 g2 = {0,0,0,0};
  i32x4 g3 = {0,0,0,0};
  i32x8 g4 = {0,0,0,0,0,0,0,0};   // 6-arg clang-23 form: extra group, zero-filled
  __builtin_amdgcn_tensor_load_to_lds(g0, g1, g2, g3, g4, 0);
}
__device__ __forceinline__ unsigned lds_off(const void* p) {
  return (unsigned)(unsigned long long)p;   // generic LDS addr: low 32 bits = LDS offset
}

// ---------------- kernel 0: weights f32 -> bf16 -------------------------------
__global__ __launch_bounds__(256) void k_cvtW(const float* __restrict__ Wq, const float* __restrict__ Wk,
                                              const float* __restrict__ Wv, const float* __restrict__ Wo,
                                              unsigned short* __restrict__ Wqb, unsigned short* __restrict__ Wkb,
                                              unsigned short* __restrict__ Wvb, unsigned short* __restrict__ Wob) {
  int i = blockIdx.x * 256 + threadIdx.x;
  if (i < KK * DD) {
    Wqb[i] = f2bf(Wq[i]); Wkb[i] = f2bf(Wk[i]);
    Wvb[i] = f2bf(Wv[i]); Wob[i] = f2bf(Wo[i]);
  }
}

// ---------------- kernel 1: fused Q/K/V projections + elu+1 -------------------
// Y[m][n] = sum_d x[m][d] * W[n][d].  Weights (3 matrices stacked: 192 rows) staged
// per-d-slice into LDS by the Tensor Data Mover, double-buffered under TENSORcnt.
__global__ __launch_bounds__(256) void k_proj(const float* __restrict__ x,
    const unsigned short* __restrict__ Wall,   // contiguous [192][1024] bf16 (Wq|Wk|Wv)
    unsigned short* __restrict__ Qb, unsigned short* __restrict__ Kb, unsigned short* __restrict__ Vb) {
  __shared__ unsigned short xt[2][128 * 32];   // x tile, bf16       (2 x 8KB)
  __shared__ unsigned short Wl[2][192 * 32];   // weight slice, bf16 (2 x 12KB)
  const int tid = threadIdx.x;
  const int wave = tid >> 5, lane = tid & 31;
  const int row0 = blockIdx.x * 128;

  // prologue: TDM weight slice 0, stage x slice 0
  if (wave == 0) tdm_load_w(Wall + 0, lds_off(&Wl[0][0]));
  {
    for (int i = tid; i < 128 * 32 / 4; i += 256) {
      int r = i >> 3, c4 = (i & 7) * 4;
      float4 f = *reinterpret_cast<const float4*>(&x[(size_t)(row0 + r) * DD + c4]);
      ushort4 h; h.x = f2bf(f.x); h.y = f2bf(f.y); h.z = f2bf(f.z); h.w = f2bf(f.w);
      *reinterpret_cast<ushort4*>(&xt[0][r * 32 + c4]) = h;
    }
  }
  if (wave == 0) __builtin_amdgcn_s_wait_tensorcnt(0);
  __syncthreads();

  v8f acc[3][4] = {};
  for (int it = 0; it < NDIT; ++it) {
    const int cur = it & 1, nxt = cur ^ 1;
    const int d0 = it * 32;
    if (it + 1 < NDIT && wave == 0)
      tdm_load_w(Wall + (d0 + 32), lds_off(&Wl[nxt][0]));   // buf nxt idle since last barrier-1

    v16bf a = load_fragA(&xt[cur][wave * 16 * 32], 32, lane);
#pragma unroll
    for (int m = 0; m < 3; ++m) {
#pragma unroll
      for (int n = 0; n < 4; ++n) {
        v16bf b = load_fragB(&Wl[cur][(m * 64 + n * 16) * 32], 32, lane);
        acc[m][n] = wmma_bf16(a, b, acc[m][n]);
      }
    }

    if (it + 1 < NDIT) {   // stage next x slice (f32 -> bf16, float4/b128 loads)
      for (int i = tid; i < 128 * 32 / 4; i += 256) {
        int r = i >> 3, c4 = (i & 7) * 4;
        float4 f = *reinterpret_cast<const float4*>(&x[(size_t)(row0 + r) * DD + d0 + 32 + c4]);
        ushort4 h; h.x = f2bf(f.x); h.y = f2bf(f.y); h.z = f2bf(f.z); h.w = f2bf(f.w);
        *reinterpret_cast<ushort4*>(&xt[nxt][r * 32 + c4]) = h;
      }
    }
    if (wave == 0) __builtin_amdgcn_s_wait_tensorcnt(0);
    __syncthreads();
  }

  const int half = lane >> 4, col16 = lane & 15;
#pragma unroll
  for (int n = 0; n < 4; ++n) {
#pragma unroll
    for (int r = 0; r < 8; ++r) {
      int row = row0 + wave * 16 + r + 8 * half;
      int col = n * 16 + col16;
      float q = acc[0][n][r]; q = q > 0.f ? q + 1.f : __expf(q);   // elu(x)+1
      float k = acc[1][n][r]; k = k > 0.f ? k + 1.f : __expf(k);
      float vv = acc[2][n][r];
      Qb[(size_t)row * KK + col] = f2bf(q);
      Kb[(size_t)row * KK + col] = f2bf(k);
      Vb[(size_t)row * KK + col] = f2bf(vv);
    }
  }
}

// ---------------- kernel 2: per-chunk sums  S_c = Kf_c^T @ V_c, ksum_c --------
__global__ __launch_bounds__(256) void k_chunksum(const unsigned short* __restrict__ Kb,
    const unsigned short* __restrict__ Vb, float* __restrict__ Ssum, float* __restrict__ ksum) {
  __shared__ unsigned short KfT[KK * CC];   // [feature][s]
  __shared__ unsigned short VT[KK * CC];    // [feature][s]
  const int b = blockIdx.y, c = blockIdx.x;
  const int tid = threadIdx.x, wave = tid >> 5, lane = tid & 31;
  const int s0 = b * SS + c * CC;
  for (int i = tid; i < CC * KK; i += 256) {
    int s = i >> 6, k = i & 63;
    KfT[k * CC + s] = Kb[(size_t)(s0 + s) * KK + k];
    VT[k * CC + s]  = Vb[(size_t)(s0 + s) * KK + k];
  }
  __syncthreads();
  const int cidx = b * NC + c;
#pragma unroll
  for (int t = 0; t < 2; ++t) {
    int ti = wave * 2 + t;
    int mi = ti >> 2, ni = ti & 3;
    v8f acc = {0,0,0,0,0,0,0,0};
#pragma unroll
    for (int sp = 0; sp < CC; sp += 32) {
      v16bf a  = load_fragA(&KfT[mi * 16 * CC + sp], CC, lane);  // A[k][s] = Kf[s][k]
      v16bf bb = load_fragB(&VT [ni * 16 * CC + sp], CC, lane);  // B[s][m] = V[s][m]
      acc = wmma_bf16(a, bb, acc);
    }
#pragma unroll
    for (int r = 0; r < 8; ++r) {
      int kr = mi * 16 + r + 8 * (lane >> 4);
      int mc = ni * 16 + (lane & 15);
      Ssum[((size_t)cidx * KK + kr) * KK + mc] = acc[r];
    }
  }
  if (tid < KK) {
    float s = 0.f;
    for (int t = 0; t < CC; ++t) s += bf2f(KfT[tid * CC + t]);
    ksum[(size_t)cidx * KK + tid] = s;
  }
}

// ---------------- kernel 3: exclusive prefix scan over chunks -----------------
__global__ __launch_bounds__(256) void k_scan(const float* __restrict__ Ssum, const float* __restrict__ ksum,
                                              float* __restrict__ Sprev, float* __restrict__ kprev) {
  const int b = blockIdx.y;
  const int e = blockIdx.x * 256 + threadIdx.x;
  if (e < KK * KK) {
    float run = 0.f;
    for (int c = 0; c < NC; ++c) {
      size_t off = (size_t)(b * NC + c) * (KK * KK) + e;
      Sprev[off] = run;
      run += Ssum[off];
    }
  } else if (e < KK * KK + KK) {
    int k = e - KK * KK;
    float run = 0.f;
    for (int c = 0; c < NC; ++c) {
      size_t off = (size_t)(b * NC + c) * KK + k;
      kprev[off] = run;
      run += ksum[off];
    }
  }
}

// ---------------- kernel 4: per-chunk attention ------------------------------
// num = Q@State_prev + mask(Q@Kf^T)@V ; den = Q.kprev + rowsum(mask(Q@Kf^T))
__global__ __launch_bounds__(256) void k_attn(const unsigned short* __restrict__ Qb,
    const unsigned short* __restrict__ Kb, const unsigned short* __restrict__ Vb,
    const float* __restrict__ Sprev, const float* __restrict__ kprev,
    unsigned short* __restrict__ Ob) {
  __shared__ unsigned short Ab[CC * CC];   // causal-masked Q@Kf^T, bf16 (32KB)
  __shared__ unsigned short VT[KK * CC];   // V transposed [m][s]    (16KB)
  __shared__ unsigned short St[KK * KK];   // StateT [m][k]          (8KB)
  __shared__ float den[CC];
  const int b = blockIdx.y, c = blockIdx.x;
  const int tid = threadIdx.x, wave = tid >> 5, lane = tid & 31;
  const int s0 = b * SS + c * CC;
  const float* Sp = Sprev + (size_t)(b * NC + c) * (KK * KK);
  const float* kp = kprev + (size_t)(b * NC + c) * KK;
  for (int i = tid; i < CC * KK; i += 256) {
    int s = i >> 6, m = i & 63;
    VT[m * CC + s] = Vb[(size_t)(s0 + s) * KK + m];
  }
  for (int i = tid; i < KK * KK; i += 256) {
    int k = i >> 6, m = i & 63;
    St[m * KK + k] = f2bf(Sp[k * KK + m]);
  }
  v16bf aq0 = load_fragA(Qb + (size_t)(s0 + wave * 16) * KK + 0,  KK, lane);
  v16bf aq1 = load_fragA(Qb + (size_t)(s0 + wave * 16) * KK + 32, KK, lane);
  const int half = lane >> 4, col16 = lane & 15;
  // GEMM1: A = Q @ Kf^T, causal mask, store bf16 to LDS. B[k][s'] = Kf[s'][k] -> Kb row-major.
#pragma unroll
  for (int nt = 0; nt < 8; ++nt) {
    v8f acc = {0,0,0,0,0,0,0,0};
    v16bf b0 = load_fragB(Kb + (size_t)(s0 + nt * 16) * KK + 0,  KK, lane);
    v16bf b1 = load_fragB(Kb + (size_t)(s0 + nt * 16) * KK + 32, KK, lane);
    acc = wmma_bf16(aq0, b0, acc);
    acc = wmma_bf16(aq1, b1, acc);
    int col = nt * 16 + col16;
#pragma unroll
    for (int r = 0; r < 8; ++r) {
      int row = wave * 16 + r + 8 * half;
      float v = (col <= row) ? acc[r] : 0.f;   // inclusive causal mask
      Ab[row * CC + col] = f2bf(v);
    }
  }
  __syncthreads();
  if (tid < CC) {
    float d = 0.f;
    for (int k = 0; k < KK; ++k) d += bf2f(Qb[(size_t)(s0 + tid) * KK + k]) * kp[k];
    for (int t = 0; t <= tid; ++t) d += bf2f(Ab[tid * CC + t]);
    den[tid] = d;
  }
  __syncthreads();
#pragma unroll
  for (int nt = 0; nt < 4; ++nt) {
    v8f acc = {0,0,0,0,0,0,0,0};
    {
      v16bf b0 = load_fragB(&St[nt * 16 * KK + 0],  KK, lane);   // B[k][m]=State[k][m]=St[m][k]
      v16bf b1 = load_fragB(&St[nt * 16 * KK + 32], KK, lane);
      acc = wmma_bf16(aq0, b0, acc);
      acc = wmma_bf16(aq1, b1, acc);
    }
#pragma unroll
    for (int sp = 0; sp < CC; sp += 32) {
      v16bf a  = load_fragA(&Ab[wave * 16 * CC + sp], CC, lane);
      v16bf bb = load_fragB(&VT[nt * 16 * CC + sp],   CC, lane); // B[s'][m]=V[s'][m]=VT[m][s']
      acc = wmma_bf16(a, bb, acc);
    }
    int col = nt * 16 + col16;
#pragma unroll
    for (int r = 0; r < 8; ++r) {
      int row = wave * 16 + r + 8 * half;
      float v = acc[r] / (den[row] + 1e-6f);
      Ob[(size_t)(s0 + row) * KK + col] = f2bf(v);
    }
  }
}

// ---------------- kernel 5: out = attn @ Wo^T --------------------------------
__global__ __launch_bounds__(256) void k_out(const unsigned short* __restrict__ Ob,
    const unsigned short* __restrict__ Wob, float* __restrict__ out) {
  const int tid = threadIdx.x, wave = tid >> 5, lane = tid & 31;
  const int row0 = blockIdx.x * 128 + wave * 16;
  const int n0 = blockIdx.y * 128;
  v16bf a0 = load_fragA(Ob + (size_t)row0 * KK + 0,  KK, lane);
  v16bf a1 = load_fragA(Ob + (size_t)row0 * KK + 32, KK, lane);
  const int half = lane >> 4, col16 = lane & 15;
#pragma unroll
  for (int nt = 0; nt < 8; ++nt) {
    v8f acc = {0,0,0,0,0,0,0,0};
    v16bf b0 = load_fragB(Wob + (size_t)(n0 + nt * 16) * KK + 0,  KK, lane);
    v16bf b1 = load_fragB(Wob + (size_t)(n0 + nt * 16) * KK + 32, KK, lane);
    acc = wmma_bf16(a0, b0, acc);
    acc = wmma_bf16(a1, b1, acc);
#pragma unroll
    for (int r = 0; r < 8; ++r) {
      int row = row0 + r + 8 * half;
      int col = n0 + nt * 16 + col16;
      out[(size_t)row * DD + col] = acc[r];
    }
  }
}

// ---------------- launch -----------------------------------------------------
extern "C" void kernel_launch(void* const* d_in, const int* in_sizes, int n_in,
                              void* d_out, int out_size, void* d_ws, size_t ws_size,
                              hipStream_t stream) {
  (void)in_sizes; (void)n_in; (void)out_size; (void)ws_size;
  const float* x  = (const float*)d_in[0];
  const float* Wq = (const float*)d_in[1];
  const float* Wk = (const float*)d_in[2];
  const float* Wv = (const float*)d_in[3];
  const float* Wo = (const float*)d_in[4];
  float* out = (float*)d_out;

  char* ws = (char*)d_ws;
  size_t off = 0;
  auto alloc = [&](size_t bytes) -> void* {
    void* p = ws + off;
    off = (off + bytes + 255) & ~(size_t)255;
    return p;
  };
  unsigned short* Qb  = (unsigned short*)alloc((size_t)BS * KK * 2);
  unsigned short* Kb  = (unsigned short*)alloc((size_t)BS * KK * 2);
  unsigned short* Vb  = (unsigned short*)alloc((size_t)BS * KK * 2);
  unsigned short* Ob  = (unsigned short*)alloc((size_t)BS * KK * 2);
  // Wq|Wk|Wv contiguous (each 131072 bytes, 256-aligned) -> one [192][1024] tensor for TDM
  unsigned short* Wqb = (unsigned short*)alloc((size_t)KK * DD * 2);
  unsigned short* Wkb = (unsigned short*)alloc((size_t)KK * DD * 2);
  unsigned short* Wvb = (unsigned short*)alloc((size_t)KK * DD * 2);
  unsigned short* Wob = (unsigned short*)alloc((size_t)DD * KK * 2);
  float* Ssum  = (float*)alloc((size_t)BB * NC * KK * KK * 4);
  float* ksum  = (float*)alloc((size_t)BB * NC * KK * 4);
  float* Sprev = (float*)alloc((size_t)BB * NC * KK * KK * 4);
  float* kprev = (float*)alloc((size_t)BB * NC * KK * 4);

  k_cvtW<<<dim3((KK * DD + 255) / 256), 256, 0, stream>>>(Wq, Wk, Wv, Wo, Wqb, Wkb, Wvb, Wob);
  k_proj<<<dim3(BS / 128), 256, 0, stream>>>(x, Wqb, Qb, Kb, Vb);
  k_chunksum<<<dim3(NC, BB), 256, 0, stream>>>(Kb, Vb, Ssum, ksum);
  k_scan<<<dim3((KK * KK + KK + 255) / 256, BB), 256, 0, stream>>>(Ssum, ksum, Sprev, kprev);
  k_attn<<<dim3(NC, BB), 256, 0, stream>>>(Qb, Kb, Vb, Sprev, kprev, Ob);
  k_out<<<dim3(BS / 128, DD / 128), 256, 0, stream>>>(Ob, Wob, out);
}
